// CrossRegionAttention_25812753449022
// MI455X (gfx1250) — matrix operand order repaired and checked by reading, not compiled, not run on hardware
//
#include <hip/hip_runtime.h>

// ---------------------------------------------------------------------------
// CrossRegionAttention fused CDNA5 (gfx1250) implementation.
//  K1 convw      : fp32 -> bf16 transposed weights (Wq,Wk,Wv,Wpi) into ws
//  K2 avgpool    : x -> x_r (B,Nr,C) fp32
//  K3/K5 rowgemm : region-path fp32 GEMMs (qr/kr/vr, proj_region)
//  K4 regattn    : 16 small fp32 attention problems (256 tokens, d=32)
//  K6 inside     : per-window fused QKV + attention + proj + residual,
//                  all matrix math via v_wmma_f32_16x16x32_bf16.
// ---------------------------------------------------------------------------

typedef __attribute__((ext_vector_type(16))) __bf16 bf16x16;
typedef __attribute__((ext_vector_type(8)))  __bf16 bf16x8;
typedef __attribute__((ext_vector_type(8)))  float  f32x8;

#define XSTR 264   // LDS row stride (elems) for 64x256 bf16 tiles (16B-aligned rows)
#define VSTR 72    // LDS row stride for Vt (256x64) and per-head P (64x64)

static __device__ __forceinline__ f32x8 wmma_bf16(bf16x16 a, bf16x16 b, f32x8 c) {
  return __builtin_amdgcn_wmma_f32_16x16x32_bf16(false, a, false, b, (short)0, c,
                                                 false, false);
}

// A fragment (M=16,K=32) from row-major [M][K] storage.
// Lane m=lane&15; lanes 0-15: K = e(<8)->e, e(>=8)->e+8 ; lanes 16-31: +8.
static __device__ __forceinline__ bf16x16 ldA(const __bf16* base, int stride, int lane) {
  const int m  = lane & 15;
  const int kb = (lane < 16) ? 0 : 8;
  const __bf16* p = base + m * stride + kb;
  bf16x8 lo = *(const bf16x8*)(p);        // K = kb .. kb+7
  bf16x8 hi = *(const bf16x8*)(p + 16);   // K = kb+16 .. kb+23
  return __builtin_shufflevector(lo, hi, 0,1,2,3,4,5,6,7,8,9,10,11,12,13,14,15);
}

// B fragment (K=32,N=16) from K-contiguous storage BT[n][k].
// Lane n=lane&15; lanes 0-15: K = e(<8)->e, e(>=8)->e+8 ; lanes 16-31: +8.
static __device__ __forceinline__ bf16x16 ldBt(const __bf16* base, int stride, int lane) {
  const int n  = lane & 15;
  const int kb = (lane < 16) ? 0 : 8;
  const __bf16* p = base + n * stride + kb;
  bf16x8 lo = *(const bf16x8*)(p);
  bf16x8 hi = *(const bf16x8*)(p + 16);
  return __builtin_shufflevector(lo, hi, 0,1,2,3,4,5,6,7,8,9,10,11,12,13,14,15);
}

// ---------------------------------------------------------------------------
// K1: transpose+convert 4 weight matrices (256x256 fp32 [in][out]) to bf16
//     WT[mat][out][in] so WMMA B-fragments are contiguous 16B loads.
__global__ __launch_bounds__(256) void convw_k(const float* __restrict__ Wq,
                                               const float* __restrict__ Wk,
                                               const float* __restrict__ Wv,
                                               const float* __restrict__ Wpi,
                                               __bf16* __restrict__ WT) {
  int idx = blockIdx.x * 256 + threadIdx.x;          // 0 .. 262143
  int mat = idx >> 16;
  int r   = idx & 65535;
  int o   = r >> 8;
  int i   = r & 255;
  const float* W = (mat == 0) ? Wq : (mat == 1) ? Wk : (mat == 2) ? Wv : Wpi;
  WT[mat * 65536 + o * 256 + i] = (__bf16)W[i * 256 + o];
}

// ---------------------------------------------------------------------------
// K2: AvgPool2d(16) -> x_r (B,Nr,C) fp32, flat [b*65536 + n*256 + c]
__global__ __launch_bounds__(256) void avgpool_k(const float* __restrict__ x,
                                                 float* __restrict__ xr) {
  int idx = blockIdx.x * 256 + threadIdx.x;          // 0 .. 131071
  int b   = idx >> 16;
  int rem = idx & 65535;
  int n   = rem >> 8;
  int c   = rem & 255;
  int i   = n >> 4, j = n & 15;
  const float* base = x + (size_t)b * 16777216 + (size_t)c * 65536
                        + (size_t)(i * 16) * 256 + j * 16;
  float s = 0.f;
#pragma unroll
  for (int u = 0; u < 16; ++u) {
    const float4* p = (const float4*)(base + u * 256);
    float4 a = p[0], bb = p[1], cc = p[2], dd = p[3];
    s += (a.x + a.y + a.z + a.w) + (bb.x + bb.y + bb.z + bb.w)
       + (cc.x + cc.y + cc.z + cc.w) + (dd.x + dd.y + dd.z + dd.w);
  }
  xr[idx] = s * (1.0f / 256.0f);
}

// ---------------------------------------------------------------------------
// K3/K5: out[row] = in[row] @ W (+ bias). One block per row, 256 rows*cols.
__global__ __launch_bounds__(256) void rowgemm_k(const float* __restrict__ in,
                                                 const float* __restrict__ W,
                                                 const float* __restrict__ bias,
                                                 float* __restrict__ out) {
  __shared__ float xrow[256];
  const int row = blockIdx.x;
  const int col = threadIdx.x;
  xrow[col] = in[row * 256 + col];
  __syncthreads();
  float acc = bias ? bias[col] : 0.0f;
#pragma unroll 8
  for (int i = 0; i < 256; ++i) acc = fmaf(xrow[i], W[i * 256 + col], acc);
  out[row * 256 + col] = acc;
}

// ---------------------------------------------------------------------------
// K4: region attention. 16 independent problems of (256 tokens, d=32), fp32.
__global__ __launch_bounds__(256) void regattn_k(const float* __restrict__ qr,
                                                 const float* __restrict__ kr,
                                                 const float* __restrict__ vr,
                                                 float* __restrict__ outp) {
  extern __shared__ float sm[];
  float* Ks = sm;            // 256*32
  float* Vs = sm + 8192;     // 256*32
  const int h   = blockIdx.x;
  const int tid = threadIdx.x;
  const float* Qp = qr + h * 8192;
  const float* Kp = kr + h * 8192;
  const float* Vp = vr + h * 8192;
#pragma unroll
  for (int i = 0; i < 32; ++i) {
    int idx = i * 256 + tid;
    Ks[idx] = Kp[idx];
    Vs[idx] = Vp[idx];
  }
  float q[32];
#pragma unroll
  for (int d4 = 0; d4 < 8; ++d4) {
    float4 t = *(const float4*)(Qp + tid * 32 + d4 * 4);
    q[d4 * 4 + 0] = t.x; q[d4 * 4 + 1] = t.y;
    q[d4 * 4 + 2] = t.z; q[d4 * 4 + 3] = t.w;
  }
  __syncthreads();
  float mx = -1e30f;
  for (int j = 0; j < 256; ++j) {
    float s = 0.f;
#pragma unroll
    for (int d = 0; d < 32; ++d) s = fmaf(q[d], Ks[j * 32 + d], s);
    mx = fmaxf(mx, s);
  }
  float acc[32];
#pragma unroll
  for (int d = 0; d < 32; ++d) acc[d] = 0.f;
  float sum = 0.f;
  for (int j = 0; j < 256; ++j) {
    float s = 0.f;
#pragma unroll
    for (int d = 0; d < 32; ++d) s = fmaf(q[d], Ks[j * 32 + d], s);
    float e = __expf(s - mx);
    sum += e;
#pragma unroll
    for (int d = 0; d < 32; ++d) acc[d] = fmaf(e, Vs[j * 32 + d], acc[d]);
  }
  float inv = 1.0f / sum;
#pragma unroll
  for (int d = 0; d < 32; ++d) outp[h * 8192 + tid * 32 + d] = acc[d] * inv;
}

// ---------------------------------------------------------------------------
// K6: fused per-window inside path. One workgroup (8 waves) per window.
__global__ __launch_bounds__(256) void inside_k(const float* __restrict__ x,
                                                const __bf16* __restrict__ WT,
                                                const float* __restrict__ ar,
                                                const float* __restrict__ bpi,
                                                float* __restrict__ out) {
  extern __shared__ char smem[];
  __bf16* Xb = (__bf16*)smem;           // 64 x XSTR  (also reused as O buffer)
  __bf16* Qs = Xb + 64 * XSTR;          // 64 x XSTR
  __bf16* Ks = Qs + 64 * XSTR;          // 64 x XSTR
  __bf16* Vt = Ks + 64 * XSTR;          // 256 x VSTR (transposed V: [c][q])
  __bf16* Pb = Vt + 256 * VSTR;         // 8 heads x 64 x VSTR

  const int tid  = threadIdx.x;
  const int lane = tid & 31;
  const int wv   = tid >> 5;
  const int ln   = lane & 15;
  const int hh   = (lane < 16) ? 0 : 1;
  const int r0   = hh * 8;

  const int wg = blockIdx.x;
  const int b  = wg >> 10;
  const int l  = wg & 1023;
  const int aw = l >> 5;
  const int ah = l & 31;
  const int reg = (aw >> 1) * 16 + (ah >> 1);
  const float* arow = ar + ((size_t)(b * 256 + reg)) * 256;
  const float* xwin = x + (size_t)b * 16777216 + (size_t)(aw * 8) * 256 + ah * 8;

  // ---- Phase 1: load window, add region term, convert to bf16 [p][c] ----
#pragma unroll
  for (int it = 0; it < 8; ++it) {
    int r  = it * 256 + tid;   // (c, kw) row id, 0..2047
    int c  = r >> 3;
    int kw = r & 7;
    const float* src = xwin + (size_t)c * 65536 + kw * 256;
    float4 f0 = *(const float4*)(src);
    float4 f1 = *(const float4*)(src + 4);
    float a = arow[c];
    __bf16* dst = Xb + (kw * 8) * XSTR + c;
    dst[0 * XSTR] = (__bf16)(f0.x + a);
    dst[1 * XSTR] = (__bf16)(f0.y + a);
    dst[2 * XSTR] = (__bf16)(f0.z + a);
    dst[3 * XSTR] = (__bf16)(f0.w + a);
    dst[4 * XSTR] = (__bf16)(f1.x + a);
    dst[5 * XSTR] = (__bf16)(f1.y + a);
    dst[6 * XSTR] = (__bf16)(f1.z + a);
    dst[7 * XSTR] = (__bf16)(f1.w + a);
  }
  __syncthreads();

  // ---- Phase 2: QKV GEMMs (64x256 @ 256x256), wave wv owns n-tiles {2wv,2wv+1} ----
  const float qscale = 0.17677669529663687f;  // dh^-0.5, dh=32
  for (int mat = 0; mat < 3; ++mat) {
    const __bf16* Wm = WT + mat * 65536;      // [out][in] bf16
    for (int ni = 0; ni < 2; ++ni) {
      const int n = wv * 2 + ni;
      f32x8 acc[4] = {{0,0,0,0,0,0,0,0},{0,0,0,0,0,0,0,0},
                      {0,0,0,0,0,0,0,0},{0,0,0,0,0,0,0,0}};
#pragma unroll
      for (int k = 0; k < 8; ++k) {
        if (k < 7) __builtin_prefetch(Wm + (n * 16) * 256 + (k + 1) * 32, 0, 1);
        bf16x16 bf = ldBt(Wm + (n * 16) * 256 + k * 32, 256, lane);
#pragma unroll
        for (int m = 0; m < 4; ++m) {
          bf16x16 af = ldA(Xb + (m * 16) * XSTR + k * 32, XSTR, lane);
          acc[m] = wmma_bf16(af, bf, acc[m]);
        }
      }
      const int col = n * 16 + ln;
      if (mat == 0) {
#pragma unroll
        for (int m = 0; m < 4; ++m)
#pragma unroll
          for (int v = 0; v < 8; ++v)
            Qs[(m * 16 + r0 + v) * XSTR + col] = (__bf16)(acc[m][v] * qscale);
      } else if (mat == 1) {
#pragma unroll
        for (int m = 0; m < 4; ++m)
#pragma unroll
          for (int v = 0; v < 8; ++v)
            Ks[(m * 16 + r0 + v) * XSTR + col] = (__bf16)acc[m][v];
      } else {  // V transposed into Vt[c][q], packed pair stores
        __bf16* vrow = Vt + col * VSTR;
#pragma unroll
        for (int m = 0; m < 4; ++m) {
          const int rb = m * 16 + r0;
#pragma unroll
          for (int v = 0; v < 8; v += 2) {
            union { __bf16 h2[2]; unsigned u; } pk;
            pk.h2[0] = (__bf16)acc[m][v];
            pk.h2[1] = (__bf16)acc[m][v + 1];
            *(unsigned*)(vrow + rb + v) = pk.u;
          }
        }
      }
    }
  }
  __syncthreads();

  // ---- Phase 3: attention; wave wv handles head h = wv (dh=32 -> one K-step) ----
  {
    const int h = wv;
    bf16x16 qa[4];
#pragma unroll
    for (int m = 0; m < 4; ++m)
      qa[m] = ldA(Qs + (m * 16) * XSTR + h * 32, XSTR, lane);

    f32x8 s[4][4];
#pragma unroll
    for (int qt = 0; qt < 4; ++qt) {
      bf16x16 kb = ldBt(Ks + (qt * 16) * XSTR + h * 32, XSTR, lane);
#pragma unroll
      for (int m = 0; m < 4; ++m) {
        f32x8 z = {0,0,0,0,0,0,0,0};
        s[m][qt] = wmma_bf16(qa[m], kb, z);
      }
    }
    // softmax over q (64 values/row, rows live in 16-lane halves)
    __bf16* ph = Pb + h * 64 * VSTR;
#pragma unroll
    for (int m = 0; m < 4; ++m) {
#pragma unroll
      for (int v = 0; v < 8; ++v) {
        float mx = fmaxf(fmaxf(s[m][0][v], s[m][1][v]),
                         fmaxf(s[m][2][v], s[m][3][v]));
#pragma unroll
        for (int off = 1; off < 16; off <<= 1) mx = fmaxf(mx, __shfl_xor(mx, off));
        float e0 = __expf(s[m][0][v] - mx);
        float e1 = __expf(s[m][1][v] - mx);
        float e2 = __expf(s[m][2][v] - mx);
        float e3 = __expf(s[m][3][v] - mx);
        float sum = e0 + e1 + e2 + e3;
#pragma unroll
        for (int off = 1; off < 16; off <<= 1) sum += __shfl_xor(sum, off);
        float inv = 1.0f / sum;
        s[m][0][v] = e0 * inv; s[m][1][v] = e1 * inv;
        s[m][2][v] = e2 * inv; s[m][3][v] = e3 * inv;
      }
#pragma unroll
      for (int qt = 0; qt < 4; ++qt)
#pragma unroll
        for (int v = 0; v < 8; ++v)
          ph[(m * 16 + r0 + v) * VSTR + qt * 16 + ln] = (__bf16)s[m][qt][v];
    }
    __syncthreads();

    // PV: (64x64) @ (64x32) -> O[p][32h..32h+32), O aliases Xb
    __bf16* Ob = Xb;
#pragma unroll
    for (int m = 0; m < 4; ++m)
#pragma unroll
      for (int nt = 0; nt < 2; ++nt) {
        f32x8 o = {0,0,0,0,0,0,0,0};
#pragma unroll
        for (int ks = 0; ks < 2; ++ks) {
          bf16x16 pa = ldA(ph + (m * 16) * VSTR + ks * 32, VSTR, lane);
          bf16x16 vb = ldBt(Vt + (h * 32 + nt * 16) * VSTR + ks * 32, VSTR, lane);
          o = wmma_bf16(pa, vb, o);
        }
        const int col = h * 32 + nt * 16 + ln;
#pragma unroll
        for (int v = 0; v < 8; ++v)
          Ob[(m * 16 + r0 + v) * XSTR + col] = (__bf16)o[v];
      }
  }
  __syncthreads();

  // ---- Phase 4: proj_inside + bias + residual, streamed to global ----
  {
    const __bf16* Wm = WT + 3 * 65536;
    const __bf16* Ob = Xb;
    for (int ni = 0; ni < 2; ++ni) {
      const int n = wv * 2 + ni;
      f32x8 acc[4] = {{0,0,0,0,0,0,0,0},{0,0,0,0,0,0,0,0},
                      {0,0,0,0,0,0,0,0},{0,0,0,0,0,0,0,0}};
#pragma unroll
      for (int k = 0; k < 8; ++k) {
        bf16x16 bf = ldBt(Wm + (n * 16) * 256 + k * 32, 256, lane);
#pragma unroll
        for (int m = 0; m < 4; ++m) {
          bf16x16 af = ldA(Ob + (m * 16) * XSTR + k * 32, XSTR, lane);
          acc[m] = wmma_bf16(af, bf, acc[m]);
        }
      }
      const int col = n * 16 + ln;
      const float bias = bpi[col];
#pragma unroll
      for (int m = 0; m < 4; ++m) {
        const int kw = 2 * m + hh;       // p = m*16 + 8*hh + v  ->  kh = v
        const float* xr = xwin + (size_t)col * 65536 + kw * 256;
        float* orow = out + (xr - x);
        float4 xa = *(const float4*)(xr);
        float4 xc = *(const float4*)(xr + 4);
        float4 o0, o1;
        o0.x = acc[m][0] + bias + xa.x;  o0.y = acc[m][1] + bias + xa.y;
        o0.z = acc[m][2] + bias + xa.z;  o0.w = acc[m][3] + bias + xa.w;
        o1.x = acc[m][4] + bias + xc.x;  o1.y = acc[m][5] + bias + xc.y;
        o1.z = acc[m][6] + bias + xc.z;  o1.w = acc[m][7] + bias + xc.w;
        *(float4*)(orow)     = o0;
        *(float4*)(orow + 4) = o1;
      }
    }
  }
}

// ---------------------------------------------------------------------------
extern "C" void kernel_launch(void* const* d_in, const int* in_sizes, int n_in,
                              void* d_out, int out_size, void* d_ws, size_t ws_size,
                              hipStream_t stream) {
  const float* x   = (const float*)d_in[0];
  const float* Wq  = (const float*)d_in[1];
  const float* Wk  = (const float*)d_in[2];
  const float* Wv  = (const float*)d_in[3];
  const float* Wqr = (const float*)d_in[4];
  const float* Wkr = (const float*)d_in[5];
  const float* Wvr = (const float*)d_in[6];
  const float* Wpi = (const float*)d_in[7];
  const float* bpi = (const float*)d_in[8];
  const float* Wpr = (const float*)d_in[9];
  const float* bpr = (const float*)d_in[10];
  float* out = (float*)d_out;

  char* ws = (char*)d_ws;
  __bf16* WT  = (__bf16*)ws;                     // 4 * 65536 bf16 = 512 KB
  float* xr   = (float*)(ws + 524288);           // 131072 f
  float* qr   = (float*)(ws + 1048576);
  float* kr   = (float*)(ws + 1572864);
  float* vr   = (float*)(ws + 2097152);
  float* ratt = (float*)(ws + 2621440);
  float* arp  = (float*)(ws + 3145728);

  convw_k<<<1024, 256, 0, stream>>>(Wq, Wk, Wv, Wpi, WT);
  avgpool_k<<<512, 256, 0, stream>>>(x, xr);
  rowgemm_k<<<512, 256, 0, stream>>>(xr, Wqr, nullptr, qr);
  rowgemm_k<<<512, 256, 0, stream>>>(xr, Wkr, nullptr, kr);
  rowgemm_k<<<512, 256, 0, stream>>>(xr, Wvr, nullptr, vr);
  regattn_k<<<16, 256, 65536, stream>>>(qr, kr, vr, ratt);
  rowgemm_k<<<512, 256, 0, stream>>>(ratt, Wpr, bpr, arp);

  const size_t lds = (size_t)(3 * 64 * XSTR + 256 * VSTR + 8 * 64 * VSTR) * 2;
  inside_k<<<2048, 256, lds, stream>>>(x, WT, arp, bpi, out);
}
